// SpanClassfy_20409684591020
// MI455X (gfx1250) — matrix operand order repaired and checked by reading, not compiled, not run on hardware
//
#include <hip/hip_runtime.h>
#include <hip/hip_bf16.h>

typedef __attribute__((ext_vector_type(16))) __bf16 v16bf;
typedef __attribute__((ext_vector_type(8)))  float  v8f;

#define B_DIM   16
#define S_LEN   512
#define H_DIM   768
#define N1      64
#define WSPAN   8
#define ROWS_TOTAL (B_DIM * S_LEN)        // 8192
#define ROWS_PER_BLOCK 128                // 8 waves * 16 rows
#define THREADS 256
#define NTILES  5                         // 4 w1 tiles + 1 w_score tile

// ---------------------------------------------------------------------------
// Kernel 1: per-token MLP logit via bf16 WMMA, with
//   (a) term_weight folded into w1 at LDS staging time:
//         relu((h*tw)@w1 + b1)@w2  ==  relu(h @ (diag(tw)w1) + b1)@w2
//   (b) hv = hidden @ w_score computed as a 5th WMMA N-tile whose B columns
//       are [ws0, ws1, 0, ..., 0].
// One wave owns a 16-row strip: GEMM 16x(64+2), K=768 in 32-chunks, 5 WMMA/chunk.
// A scheduling fence pins all B-fragment ds_loads before the WMMA block so they
// clause together under a single s_wait_dscnt and the WMMAs issue back-to-back,
// while global loads remain free to prefetch the next chunk.
// ---------------------------------------------------------------------------
__global__ __launch_bounds__(THREADS) void spanc_logit_hv_kernel(
    const float* __restrict__ hidden,
    const float* __restrict__ term_weight,
    const float* __restrict__ w1,
    const float* __restrict__ b1,
    const float* __restrict__ w2,
    const float* __restrict__ b2,
    const float* __restrict__ w_score,
    float* __restrict__ tok_logit,
    float* __restrict__ hv)
{
    extern __shared__ char smem[];
    __bf16* w1t = (__bf16*)smem;                      // [64][768] bf16: tw[k]*w1[k][n], transposed
    __bf16* wsb = w1t + (size_t)N1 * H_DIM;           // [16][768] bf16: cols 0/1 = w_score, rest 0
    float*  b1l = (float*)(wsb + (size_t)16 * H_DIM); // [64]
    float*  w2l = b1l + N1;                           // [64]

    const int tid = threadIdx.x;

    // Stage w1 with term_weight folded, transposed for contiguous-K reads.
    for (int i = tid; i < H_DIM * N1; i += THREADS) {
        const int k = i >> 6, n = i & (N1 - 1);
        w1t[n * H_DIM + k] = (__bf16)(w1[i] * term_weight[k]);
    }
    // Stage the w_score B tile (cols 0/1 live, 2..15 zero).
    for (int i = tid; i < 16 * H_DIM; i += THREADS) {
        const int col = i / H_DIM, k = i - col * H_DIM;
        const float v = (col == 0) ? w_score[2 * k]
                      : (col == 1) ? w_score[2 * k + 1] : 0.f;
        wsb[i] = (__bf16)v;
    }
    if (tid < N1) { b1l[tid] = b1[tid]; w2l[tid] = w2[tid]; }
    __syncthreads();

    const int wave = tid >> 5;
    const int lane = tid & 31;
    const int ml   = lane & 15;     // row within strip (A) / column (B,C)
    const int hi   = lane >> 4;     // half-wave selector for K ranges
    const int row_base = blockIdx.x * ROWS_PER_BLOCK + wave * 16;
    const float* rowp = hidden + (size_t)(row_base + ml) * H_DIM;

    // Per-lane B-fragment base pointers (column fixed per lane).
    const __bf16* bfp[NTILES];
    #pragma unroll
    for (int t = 0; t < 4; ++t) bfp[t] = w1t + (ml + 16 * t) * H_DIM;
    bfp[4] = wsb + ml * H_DIM;

    const v8f vzero = {0.f, 0.f, 0.f, 0.f, 0.f, 0.f, 0.f, 0.f};
    v8f acc[NTILES];
    #pragma unroll
    for (int t = 0; t < NTILES; ++t) acc[t] = vzero;

    #pragma unroll 2
    for (int kc = 0; kc < H_DIM; kc += 32) {
        // A fragment: lanes 0-15 hold K = kc+{0..15},
        // lanes 16-31 hold K = kc+{16..31}, row = lane&15.
        const int ka0 = kc + hi * 16;
        const int ka1 = ka0 + 8;
        const float4 x0 = *(const float4*)(rowp + ka0);
        const float4 x1 = *(const float4*)(rowp + ka0 + 4);
        const float4 x2 = *(const float4*)(rowp + ka1);
        const float4 x3 = *(const float4*)(rowp + ka1 + 4);

        // Issue all 5 B-fragment LDS loads up front.
        // B layout: lanes 0-15 hold K = kc..kc+15, lanes 16-31 K = kc+16..kc+31.
        const int kb = kc + hi * 16;
        v16bf bf0 = *(const v16bf*)(bfp[0] + kb);
        v16bf bf1 = *(const v16bf*)(bfp[1] + kb);
        v16bf bf2 = *(const v16bf*)(bfp[2] + kb);
        v16bf bf3 = *(const v16bf*)(bfp[3] + kb);
        v16bf bf4 = *(const v16bf*)(bfp[4] + kb);

        // Scheduling fence: memory ops above cannot sink below this point, so
        // the ds_loads batch ahead of the WMMA block (one dscnt wait total).
        // WMMAs/cvts are not memory ops, so the next iteration's global loads
        // can still be hoisted up for prefetch.
        asm volatile("" ::: "memory");

        v16bf af;
        af[0]  = (__bf16)x0.x; af[1]  = (__bf16)x0.y;
        af[2]  = (__bf16)x0.z; af[3]  = (__bf16)x0.w;
        af[4]  = (__bf16)x1.x; af[5]  = (__bf16)x1.y;
        af[6]  = (__bf16)x1.z; af[7]  = (__bf16)x1.w;
        af[8]  = (__bf16)x2.x; af[9]  = (__bf16)x2.y;
        af[10] = (__bf16)x2.z; af[11] = (__bf16)x2.w;
        af[12] = (__bf16)x3.x; af[13] = (__bf16)x3.y;
        af[14] = (__bf16)x3.z; af[15] = (__bf16)x3.w;

        acc[0] = __builtin_amdgcn_wmma_f32_16x16x32_bf16(
            false, af, false, bf0, (short)0, acc[0], false, false);
        acc[1] = __builtin_amdgcn_wmma_f32_16x16x32_bf16(
            false, af, false, bf1, (short)0, acc[1], false, false);
        acc[2] = __builtin_amdgcn_wmma_f32_16x16x32_bf16(
            false, af, false, bf2, (short)0, acc[2], false, false);
        acc[3] = __builtin_amdgcn_wmma_f32_16x16x32_bf16(
            false, af, false, bf3, (short)0, acc[3], false, false);
        acc[4] = __builtin_amdgcn_wmma_f32_16x16x32_bf16(
            false, af, false, bf4, (short)0, acc[4], false, false);
    }

    // hv from accumulator 4: C layout => VGPR v, lanes 0-15 -> row v, N = lane&15;
    // lanes 16-31 -> row v+8. Columns 0/1 hold hv0/hv1.
    if (ml < 2) {
        #pragma unroll
        for (int v = 0; v < 8; ++v) {
            hv[2 * (row_base + v + hi * 8) + ml] = acc[4][v];
        }
    }

    // Logit epilogue: relu(acc + b1) * w2, reduced over N.
    float rs[8];
    #pragma unroll
    for (int v = 0; v < 8; ++v) rs[v] = 0.f;
    #pragma unroll
    for (int t = 0; t < 4; ++t) {
        const int n = ml + 16 * t;
        const float b1v = b1l[n];
        const float w2v = w2l[n];
        #pragma unroll
        for (int v = 0; v < 8; ++v) {
            const float x = acc[t][v] + b1v;
            rs[v] += fmaxf(x, 0.f) * w2v;
        }
    }
    #pragma unroll
    for (int m = 1; m < 16; m <<= 1) {
        #pragma unroll
        for (int v = 0; v < 8; ++v) rs[v] += __shfl_xor(rs[v], m, 32);
    }
    if (ml == 0) {
        const float b2v = b2[0];
        #pragma unroll
        for (int v = 0; v < 8; ++v) {
            tok_logit[row_base + v + hi * 8] = rs[v] + b2v;
        }
    }
}

// ---------------------------------------------------------------------------
// Kernel 2: per-token prefix softmax over W=8 window, scores = attn @ hv + b.
// ---------------------------------------------------------------------------
__global__ __launch_bounds__(THREADS) void spanc_attn_kernel(
    const float* __restrict__ tok_logit,
    const float* __restrict__ hv,
    const float* __restrict__ b_score,
    float* __restrict__ out)
{
    const int g = blockIdx.x * THREADS + threadIdx.x;   // 0..8191 (b*S + s)
    const int s = g & (S_LEN - 1);
    const int base = g - s;

    float l[WSPAN], h0[WSPAN], h1[WSPAN];
    #pragma unroll
    for (int j = 0; j < WSPAN; ++j) {
        const int idx = base + min(s + j, S_LEN - 1);
        l[j]  = tok_logit[idx];
        h0[j] = hv[2 * idx + 0];
        h1[j] = hv[2 * idx + 1];
    }
    const float bs0 = b_score[0], bs1 = b_score[1];

    // Online prefix softmax: span w uses positions j < min(w+1, S - s).
    float m = -1e30f, den = 0.f, n0 = 0.f, n1 = 0.f;
    #pragma unroll
    for (int w = 0; w < WSPAN; ++w) {
        if (w < S_LEN - s) {
            const float lw = l[w];
            const float nm = fmaxf(m, lw);
            const float sc = __expf(m - nm);
            const float e  = __expf(lw - nm);
            den = den * sc + e;
            n0  = n0  * sc + e * h0[w];
            n1  = n1  * sc + e * h1[w];
            m = nm;
        }
        const float inv = 1.f / den;
        out[((size_t)g * WSPAN + w) * 2 + 0] = n0 * inv + bs0;
        out[((size_t)g * WSPAN + w) * 2 + 1] = n1 * inv + bs1;
    }
}

extern "C" void kernel_launch(void* const* d_in, const int* in_sizes, int n_in,
                              void* d_out, int out_size, void* d_ws, size_t ws_size,
                              hipStream_t stream) {
    const float* hidden      = (const float*)d_in[0];
    const float* term_weight = (const float*)d_in[1];
    const float* w1          = (const float*)d_in[2];
    const float* b1          = (const float*)d_in[3];
    const float* w2          = (const float*)d_in[4];
    const float* b2          = (const float*)d_in[5];
    const float* w_score     = (const float*)d_in[6];
    const float* b_score     = (const float*)d_in[7];
    float* out = (float*)d_out;

    float* tok_logit = (float*)d_ws;            // 8192 floats
    float* hvbuf     = tok_logit + ROWS_TOTAL;  // 16384 floats

    const size_t shmem = (size_t)N1 * H_DIM * 2      // w1t bf16 (tw folded)
                       + (size_t)16 * H_DIM * 2      // w_score B tile bf16
                       + (size_t)(2 * N1) * 4;       // b1, w2

    spanc_logit_hv_kernel<<<ROWS_TOTAL / ROWS_PER_BLOCK, THREADS, shmem, stream>>>(
        hidden, term_weight, w1, b1, w2, b2, w_score, tok_logit, hvbuf);

    spanc_attn_kernel<<<ROWS_TOTAL / THREADS, THREADS, 0, stream>>>(
        tok_logit, hvbuf, b_score, out);
}
